// ScaledDotProductAttention_9113920602600
// MI455X (gfx1250) — matrix operand order repaired and checked by reading, not compiled, not run on hardware
//
#include <hip/hip_runtime.h>

typedef __attribute__((ext_vector_type(16))) _Float16 v16h;
typedef __attribute__((ext_vector_type(8)))  float    v8f;
typedef __attribute__((ext_vector_type(4)))  float    v4f;

#define BATCH 2
#define HEADS 16
#define SEQ   2048
#define DIM   64
#define QT    16                 // query rows per block
#define PSTR  (SEQ + 4)          // LDS row stride in floats (pad to dodge bank conflicts)
#define NWAVE 4
#define TPB   (NWAVE * 32)

__global__ __launch_bounds__(TPB)
void fa_fwd_attn_kernel(const float* __restrict__ Q,
                        const float* __restrict__ K,
                        const float* __restrict__ V,
                        float* __restrict__ out,
                        float* __restrict__ attn)
{
    extern __shared__ float P[];          // QT x PSTR fp32 score panel (131,328 B)
    __shared__ float red[QT][8];
    __shared__ float rowv[QT];            // per-row max, then per-row sum

    const int tid  = threadIdx.x;
    const int wid  = tid >> 5;
    const int lane = tid & 31;
    const int hi   = lane & 16;           // 0 or 16: which half-wave
    const int nl   = lane & 15;

    const int  i     = blockIdx.x;        // query tile index
    const int  bh    = blockIdx.y;        // fused batch*head
    const int  qi    = i * QT;
    const long base  = (long)bh * SEQ * DIM;
    const long abase = (long)bh * SEQ * SEQ;
    const int  L     = (i + 1) * QT;      // causal: valid key count for this tile
    const int  Lp    = (L + 31) & ~31;    // padded up to a multiple of 32 for PV GEMM

    // ---------------- Q tile -> fp16 WMMA A-layout registers ----------------
    // 16-bit A 16x32: lanes 0-15 hold K(0,1),(2,3),(4,5),(6,7),(16,17).. ; lanes 16-31 +8
    const int arow  = nl;                 // M row this lane supplies
    const int akoff = hi ? 8 : 0;
    v16h qa[2];
    for (int kc = 0; kc < 2; ++kc) {
        const float* qr = Q + base + (long)(qi + arow) * DIM + kc * 32;
        for (int v = 0; v < 8; ++v) {
            int kk = ((v < 4) ? (v * 2) : (16 + (v - 4) * 2)) + akoff;
            qa[kc][2 * v]     = (_Float16)qr[kk];
            qa[kc][2 * v + 1] = (_Float16)qr[kk + 1];
        }
    }

    // ---------------- S = scale * Q K^T with causal mask -> LDS ----------------
    // 16-bit B 32x16: lanes 0-15 hold K=0..7,16..23 of column N=lane, lanes 16-31 hold K=8..15,24..31
    const int bkoff = hi ? 8 : 0;
    for (int j = wid; j <= i; j += NWAVE) {
        v8f c = {};
        for (int kc = 0; kc < 2; ++kc) {
            const float* kr = K + base + (long)(j * 16 + nl) * DIM + kc * 32 + bkoff;
            v16h kb;
            for (int t = 0; t < 16; ++t) kb[t] = (_Float16)kr[(t < 8) ? t : (t + 8)];
            c = __builtin_amdgcn_wmma_f32_16x16x32_f16(false, qa[kc], false, kb,
                                                       (short)0, c, false, false);
        }
        // C layout: VGPR r holds row (r + (hi?8:0)), column = lane&15
        const int mb = hi ? 8 : 0;
        for (int r = 0; r < 8; ++r) {
            int   qg = qi + r + mb;
            int   kg = j * 16 + nl;
            float s  = c[r] * 0.125f;                 // 1/sqrt(64)
            if (kg > qg) s = -__builtin_inff();       // causal mask
            P[(r + mb) * PSTR + kg] = s;
        }
    }
    __syncthreads();

    // ---------------- row softmax stats; keep unnormalized exp() in LDS ----------------
    {
        const int row = tid >> 3;         // 16 rows x 8 threads each
        const int sub = tid & 7;

        float m = -__builtin_inff();
        for (int cix = sub; cix < L; cix += 8)
            m = fmaxf(m, P[row * PSTR + cix]);
        red[row][sub] = m;
        __syncthreads();
        if (sub == 0) {
            float mm = red[row][0];
            for (int t = 1; t < 8; ++t) mm = fmaxf(mm, red[row][t]);
            rowv[row] = mm;
        }
        __syncthreads();

        const float rmax = rowv[row];
        float ssum = 0.f;
        for (int cix = sub; cix < L; cix += 8) {
            float e = __expf(P[row * PSTR + cix] - rmax);
            P[row * PSTR + cix] = e;
            ssum += e;
        }
        red[row][sub] = ssum;
        __syncthreads();
        if (sub == 0) {
            float ss = red[row][0];
            for (int t = 1; t < 8; ++t) ss += red[row][t];
            rowv[row] = ss;               // row sum (always > 0: diagonal key unmasked)
        }
        // zero-pad columns [L, Lp) so PV GEMM can run K in clean chunks of 32
        if (Lp != L) {
            for (int t = tid; t < QT * 16; t += TPB) {
                int rr = t >> 4;
                P[rr * PSTR + L + (t & 15)] = 0.f;
            }
        }
    }
    __syncthreads();

    // ---------------- write normalized attn (and upper-triangle zeros) ----------------
    {
        const int row4 = SEQ / 4;         // 512 float4 per row
        for (int t = tid; t < QT * row4; t += TPB) {
            int row = t >> 9;             // t / 512
            int c4  = (t & (row4 - 1)) << 2;
            v4f v;
            if (c4 < L) {
                v = *(const v4f*)&P[row * PSTR + c4];
                float rv = 1.0f / rowv[row];
                v *= rv;
            } else {
                v = (v4f){0.f, 0.f, 0.f, 0.f};
            }
            __builtin_nontemporal_store(
                v, (v4f*)(attn + abase + (long)(qi + row) * SEQ + c4));
        }
    }

    // ---------------- O = P V : each wave owns a 16-wide slab of D ----------------
    {
        const int n0 = wid * 16;
        v8f acc = {};
        for (int kc = 0; kc < (Lp >> 5); ++kc) {
            v16h pa, vb;
            const float* pr = P + arow * PSTR + kc * 32;
            for (int v = 0; v < 8; ++v) {
                int kk = ((v < 4) ? (v * 2) : (16 + (v - 4) * 2)) + akoff;
                pa[2 * v]     = (_Float16)pr[kk];
                pa[2 * v + 1] = (_Float16)pr[kk + 1];
            }
            const float* vr = V + base + (long)(kc * 32 + bkoff) * DIM + n0 + nl;
            for (int t = 0; t < 16; ++t) vb[t] = (_Float16)vr[((t < 8) ? t : (t + 8)) * DIM];
            acc = __builtin_amdgcn_wmma_f32_16x16x32_f16(false, pa, false, vb,
                                                         (short)0, acc, false, false);
        }
        const int mb = hi ? 8 : 0;
        for (int r = 0; r < 8; ++r) {
            float rv = 1.0f / rowv[r + mb];
            out[base + (long)(qi + r + mb) * DIM + n0 + nl] = acc[r] * rv;
        }
    }
}

extern "C" void kernel_launch(void* const* d_in, const int* in_sizes, int n_in,
                              void* d_out, int out_size, void* d_ws, size_t ws_size,
                              hipStream_t stream) {
    const float* Q = (const float*)d_in[0];
    const float* K = (const float*)d_in[1];
    const float* V = (const float*)d_in[2];
    // d_in[3] is the causal mask; causality is applied analytically in-kernel.

    float* out  = (float*)d_out;
    float* attn = (float*)d_out + (size_t)BATCH * HEADS * SEQ * DIM;

    const int lds_bytes = QT * PSTR * (int)sizeof(float);   // 131,328 B (< 320 KB WGP LDS)
    (void)hipFuncSetAttribute(reinterpret_cast<const void*>(fa_fwd_attn_kernel),
                              hipFuncAttributeMaxDynamicSharedMemorySize, lds_bytes);

    dim3 grid(SEQ / QT, BATCH * HEADS);   // 128 query tiles x 32 (batch*head)
    fa_fwd_attn_kernel<<<grid, TPB, lds_bytes, stream>>>(Q, K, V, out, attn);
}
